// NeuralSDEEncoder_55679956025724
// MI455X (gfx1250) — matrix operand order repaired and checked by reading, not compiled, not run on hardware
//
#include <hip/hip_runtime.h>
#include <hip/hip_bf16.h>

// ---------------------------------------------------------------------------
// NeuralSDEEncoder, fully fused for gfx1250 (CDNA5, wave32, WMMA).
//
// Two-phase launch:
//   1) pack_weights_kernel: converts every weight matrix f32 -> f16 ONCE into
//      d_ws, in exact WMMA B-fragment order [kt][nt][lane][16], zero-padded in
//      K (65 -> 96 for the [t|z] GEMMs).  ~1.4 MB, L2-resident.
//   2) nsde_fused_kernel: one workgroup = 64 batch rows = 2 waves x 32 rows
//      (two 16-row M-tiles per wave -> each packed-B b128 load feeds TWO
//      v_wmma_f32_16x16x32_f16, halving redundant L2 weight traffic).
//      Latent z lives in VGPRs across the whole 63-step scan; activations
//      live in LDS as f16.  All LDS data is wave-private -> zero barriers in
//      the scan loop.
// ---------------------------------------------------------------------------

typedef __attribute__((ext_vector_type(16))) _Float16 v16h;
typedef __attribute__((ext_vector_type(8)))  _Float16 v8h;
typedef __attribute__((ext_vector_type(8)))  float    v8f;

constexpr int NSDE_BATCH = 16384;
constexpr int NSDE_DIN   = 256;
constexpr int NSDE_HID   = 256;
constexpr int NSDE_LAT   = 64;
constexpr int NSDE_T     = 64;
constexpr int NSDE_NBLK  = 3;

constexpr int ROWS     = 64;   // batch rows per workgroup
constexpr int NTHREADS = 64;   // 2 waves
constexpr int MROWS    = 32;   // rows per wave (2 x 16-row M-tiles)
constexpr int TZLD     = 96;   // padded K for [t|z] (65 -> 3 k-tiles of 32)

#define ACT_NONE  0
#define ACT_SWISH 1

__device__ __forceinline__ float swish_f(float v) {
  return v * (1.0f / (1.0f + __expf(-v)));
}

__global__ void pack_weights_kernel(const float* __restrict__ W,
                                    _Float16* __restrict__ dst,
                                    int K, int N, int Ktiles) {
  const int Ntiles = N >> 4;
  const size_t total = (size_t)Ktiles * Ntiles * 32 * 16;
  for (size_t idx = (size_t)blockIdx.x * blockDim.x + threadIdx.x; idx < total;
       idx += (size_t)gridDim.x * blockDim.x) {
    const int j       = (int)(idx & 15);
    const int lane    = (int)((idx >> 4) & 31);
    const size_t tile = idx >> 9;               // kt*Ntiles + nt
    const int nt = (int)(tile % Ntiles);
    const int kt = (int)(tile / Ntiles);
    const int k = kt * 32 + 8 * (lane >> 4) + (j & 7) + 16 * (j >> 3);
    const int n = nt * 16 + (lane & 15);
    dst[idx] = (k < K) ? (_Float16)W[(size_t)k * N + n] : (_Float16)0.0f;
  }
}

// Accumulate TWO 16x16 output tiles (M-tiles rowbase, rowbase+16); one packed
// B-fragment b128 load is shared by both WMMAs.
template <int KT>
__device__ __forceinline__ void wmma_acc_mt2(
    const _Float16* __restrict__ Alds, int lda, int rowbase,
    const _Float16* __restrict__ Wpk, int Ntiles, int nt,
    float bias, int lane, v8f acc[2])
{
  const int half = lane >> 4;
  const int ml   = lane & 15;
#pragma unroll
  for (int mt = 0; mt < 2; ++mt)
#pragma unroll
    for (int i = 0; i < 8; ++i) acc[mt][i] = bias;

  const _Float16* __restrict__ arow0 = Alds + (size_t)(rowbase + ml) * lda;
  const _Float16* __restrict__ arow1 = arow0 + (size_t)16 * lda;
  const _Float16* __restrict__ wp    = Wpk + ((size_t)nt * 32 + lane) * 16;
  const size_t wstep = (size_t)Ntiles * 512;   // halfs per k-tile of packed W

#pragma unroll
  for (int kt = 0; kt < KT; ++kt) {
    const int kk = kt * 32;
    const v16h b = *(const v16h*)(wp + (size_t)kt * wstep);

    const v8h a0lo = *(const v8h*)(arow0 + kk + 8 * half);
    const v8h a0hi = *(const v8h*)(arow0 + kk + 16 + 8 * half);
    const v8h a1lo = *(const v8h*)(arow1 + kk + 8 * half);
    const v8h a1hi = *(const v8h*)(arow1 + kk + 16 + 8 * half);
    v16h a0, a1;
#pragma unroll
    for (int i = 0; i < 8; ++i) {
      a0[i] = a0lo[i]; a0[8 + i] = a0hi[i];
      a1[i] = a1lo[i]; a1[8 + i] = a1hi[i];
    }

    acc[0] = __builtin_amdgcn_wmma_f32_16x16x32_f16(
        false, a0, false, b, (short)0, acc[0], false, false);
    acc[1] = __builtin_amdgcn_wmma_f32_16x16x32_f16(
        false, a1, false, b, (short)0, acc[1], false, false);
  }
}

// One full layer for this wave's 32 rows.  resid may alias O (in-place).
template <int KT>
__device__ __forceinline__ void gemm_layer(
    const _Float16* __restrict__ A, int lda,
    const _Float16* __restrict__ Wpk, const float* __restrict__ bias, int Ntiles,
    _Float16* O, int ldo,
    const _Float16* resid, int ldr,
    int rowbase, int lane, int act)
{
  const int half = lane >> 4;
  const int ml   = lane & 15;
  for (int nt = 0; nt < Ntiles; ++nt) {
    const int n0 = nt * 16;
    v8f acc[2];
    wmma_acc_mt2<KT>(A, lda, rowbase, Wpk, Ntiles, nt, bias[n0 + ml], lane, acc);
#pragma unroll
    for (int mt = 0; mt < 2; ++mt) {
#pragma unroll
      for (int i = 0; i < 8; ++i) {
        float v = acc[mt][i];
        const int r = rowbase + mt * 16 + 8 * half + i;
        if (resid) v += (float)resid[(size_t)r * ldr + n0 + ml];
        if (act == ACT_SWISH) v = swish_f(v);
        O[(size_t)r * ldo + n0 + ml] = (_Float16)v;
      }
    }
  }
}

__global__ void __launch_bounds__(NTHREADS, 1)
nsde_fused_kernel(
    const float* __restrict__ x, const float* __restrict__ ts,
    const float* __restrict__ noise,
    const _Float16* __restrict__ Wp_in,  const float* __restrict__ b_in,
    const _Float16* __restrict__ Wp_b1,  const float* __restrict__ bb1,
    const _Float16* __restrict__ Wp_b2,  const float* __restrict__ bb2,
    const _Float16* __restrict__ Wp_fo,  const float* __restrict__ b_fo,
    const _Float16* __restrict__ Wp_lat, const float* __restrict__ b_lat,
    const _Float16* __restrict__ Wp_init,const float* __restrict__ b_init,
    const _Float16* __restrict__ Wp_d1,  const float* __restrict__ bd1,
    const _Float16* __restrict__ Wp_d2,  const float* __restrict__ bd2,
    const _Float16* __restrict__ Wp_d3,  const float* __restrict__ bd3,
    const _Float16* __restrict__ Wp_g1,  const float* __restrict__ bg1,
    const _Float16* __restrict__ Wp_g2,  const float* __restrict__ bg2,
    float* __restrict__ out_mu, float* __restrict__ out_s,
    float* __restrict__ out_zpath)
{
  extern __shared__ _Float16 smem[];
  _Float16* bufA = smem;                         // [ROWS][HID]
  _Float16* bufB = bufA + ROWS * NSDE_HID;       // [ROWS][HID]
  _Float16* tzb  = bufB + ROWS * NSDE_HID;       // [ROWS][TZLD]  ([t|z|0-pad])

  const int tid     = threadIdx.x;
  const int lane    = tid & 31;
  const int wave    = tid >> 5;
  const int rowbase = wave * MROWS;
  const int half    = lane >> 4;
  const int ml      = lane & 15;
  const int tile0   = blockIdx.x * ROWS;

  for (int i = tid; i < ROWS * TZLD; i += NTHREADS) tzb[i] = (_Float16)0.0f;
  for (int i = tid; i < ROWS * NSDE_DIN; i += NTHREADS) {
    const int r = i >> 8, c = i & (NSDE_DIN - 1);
    bufA[i] = (_Float16)x[(size_t)(tile0 + r) * NSDE_DIN + c];
  }
  __syncthreads();   // only barrier: all later LDS accesses are wave-private

  // ---------------- front-end MLP (2 buffers, in-place residual) ----------
  gemm_layer<8>(bufA, NSDE_HID, Wp_in, b_in, 16,
                bufB, NSDE_HID, nullptr, 0, rowbase, lane, ACT_SWISH);
  for (int blk = 0; blk < NSDE_NBLK; ++blk) {
    const _Float16* w1 = Wp_b1 + (size_t)blk * 65536;
    const _Float16* w2 = Wp_b2 + (size_t)blk * 65536;
    gemm_layer<8>(bufB, NSDE_HID, w1, bb1 + blk * NSDE_HID, 16,
                  bufA, NSDE_HID, nullptr, 0, rowbase, lane, ACT_SWISH);
    gemm_layer<8>(bufA, NSDE_HID, w2, bb2 + blk * NSDE_HID, 16,
                  bufB, NSDE_HID, /*resid=*/bufB, NSDE_HID,
                  rowbase, lane, ACT_NONE);
  }
  gemm_layer<8>(bufB, NSDE_HID, Wp_fo, b_fo, 16,
                bufA, NSDE_HID, nullptr, 0, rowbase, lane, ACT_NONE);

  // ---------------- latent params: mu, s = exp(0.5*logvar) ----------------
  for (int nt = 0; nt < 8; ++nt) {
    const int n0 = nt * 16;
    v8f acc[2];
    wmma_acc_mt2<8>(bufA, NSDE_HID, rowbase, Wp_lat, 8, nt,
                    b_lat[n0 + ml], lane, acc);
#pragma unroll
    for (int mt = 0; mt < 2; ++mt)
#pragma unroll
      for (int i = 0; i < 8; ++i) {
        const int row = tile0 + rowbase + mt * 16 + 8 * half + i;
        const int c   = n0 + ml;
        if (c < NSDE_LAT)
          out_mu[(size_t)row * NSDE_LAT + c] = acc[mt][i];
        else
          out_s[(size_t)row * NSDE_LAT + (c - NSDE_LAT)] =
              __expf(0.5f * acc[mt][i]);
      }
  }

  // ---------------- z0 = features @ W_init + b_init (kept in VGPRs) -------
  v8f z[2][4];
#pragma unroll
  for (int nt = 0; nt < 4; ++nt) {
    v8f acc[2];
    wmma_acc_mt2<8>(bufA, NSDE_HID, rowbase, Wp_init, 4, nt,
                    b_init[nt * 16 + ml], lane, acc);
    z[0][nt] = acc[0];
    z[1][nt] = acc[1];
#pragma unroll
    for (int mt = 0; mt < 2; ++mt)
#pragma unroll
      for (int i = 0; i < 8; ++i) {
        const int row = tile0 + rowbase + mt * 16 + 8 * half + i;
        out_zpath[(size_t)row * NSDE_LAT + nt * 16 + ml] = acc[mt][i]; // step 0
      }
  }

  // ---------------- Euler-Maruyama scan (63 steps, barrier-free) ----------
  for (int st = 0; st < NSDE_T - 1; ++st) {
    const float t    = ts[st];
    const float dt   = ts[st + 1] - t;
    const float sqdt = __fsqrt_rn(dt);

    tzb[(size_t)tid * TZLD] = (_Float16)t;       // own row (64 thr / 64 rows)
#pragma unroll
    for (int mt = 0; mt < 2; ++mt)
#pragma unroll
      for (int nt = 0; nt < 4; ++nt)
#pragma unroll
        for (int i = 0; i < 8; ++i)
          tzb[(size_t)(rowbase + mt * 16 + 8 * half + i) * TZLD
              + 1 + nt * 16 + ml] = (_Float16)z[mt][nt][i];

    // drift MLP
    gemm_layer<3>(tzb, TZLD, Wp_d1, bd1, 16,
                  bufB, NSDE_HID, nullptr, 0, rowbase, lane, ACT_SWISH);
    gemm_layer<8>(bufB, NSDE_HID, Wp_d2, bd2, 16,
                  bufA, NSDE_HID, nullptr, 0, rowbase, lane, ACT_SWISH);
    v8f dr[2][4];
#pragma unroll
    for (int nt = 0; nt < 4; ++nt) {
      v8f acc[2];
      wmma_acc_mt2<8>(bufA, NSDE_HID, rowbase, Wp_d3, 4, nt,
                      bd3[nt * 16 + ml], lane, acc);
      dr[0][nt] = acc[0];
      dr[1][nt] = acc[1];
    }

    // diffusion MLP + z update + z_path store
    gemm_layer<3>(tzb, TZLD, Wp_g1, bg1, 16,
                  bufB, NSDE_HID, nullptr, 0, rowbase, lane, ACT_SWISH);
#pragma unroll
    for (int nt = 0; nt < 4; ++nt) {
      v8f acc[2];
      wmma_acc_mt2<8>(bufB, NSDE_HID, rowbase, Wp_g2, 4, nt,
                      bg2[nt * 16 + ml], lane, acc);
#pragma unroll
      for (int mt = 0; mt < 2; ++mt)
#pragma unroll
        for (int i = 0; i < 8; ++i) {
          const float g   = __logf(1.0f + __expf(acc[mt][i]));   // softplus
          const int  row  = tile0 + rowbase + mt * 16 + 8 * half + i;
          const int  c    = nt * 16 + ml;
          const float nz  = noise[((size_t)st * NSDE_BATCH + row) * NSDE_LAT + c];
          const float znew = z[mt][nt][i] + dr[mt][nt][i] * dt + g * sqdt * nz;
          z[mt][nt][i] = znew;
          out_zpath[((size_t)(st + 1) * NSDE_BATCH + row) * NSDE_LAT + c] = znew;
        }
    }
  }
}

extern "C" void kernel_launch(void* const* d_in, const int* in_sizes, int n_in,
                              void* d_out, int out_size, void* d_ws, size_t ws_size,
                              hipStream_t stream) {
  (void)in_sizes; (void)n_in; (void)out_size; (void)ws_size;
  const float* x      = (const float*)d_in[0];
  const float* ts     = (const float*)d_in[1];
  const float* noise  = (const float*)d_in[2];
  const float* W_in   = (const float*)d_in[3];
  const float* b_in   = (const float*)d_in[4];
  const float* Wb1    = (const float*)d_in[5];
  const float* bb1    = (const float*)d_in[6];
  const float* Wb2    = (const float*)d_in[7];
  const float* bb2    = (const float*)d_in[8];
  const float* W_fo   = (const float*)d_in[9];
  const float* b_fo   = (const float*)d_in[10];
  const float* W_lat  = (const float*)d_in[11];
  const float* b_lat  = (const float*)d_in[12];
  const float* W_init = (const float*)d_in[13];
  const float* b_init = (const float*)d_in[14];
  const float* Wd1    = (const float*)d_in[15];
  const float* bd1    = (const float*)d_in[16];
  const float* Wd2    = (const float*)d_in[17];
  const float* bd2    = (const float*)d_in[18];
  const float* Wd3    = (const float*)d_in[19];
  const float* bd3    = (const float*)d_in[20];
  const float* Wg1    = (const float*)d_in[21];
  const float* bg1    = (const float*)d_in[22];
  const float* Wg2    = (const float*)d_in[23];
  const float* bg2    = (const float*)d_in[24];

  _Float16* wsh = (_Float16*)d_ws;
  size_t off = 0;
  auto carve = [&](int Ktiles, int N) {
    _Float16* p = wsh + off;
    off += (size_t)Ktiles * (N / 16) * 512;
    return p;
  };
  _Float16* Wp_in   = carve(8, 256);
  _Float16* Wp_b1   = carve(8, 256); carve(8, 256); carve(8, 256);
  _Float16* Wp_b2   = carve(8, 256); carve(8, 256); carve(8, 256);
  _Float16* Wp_fo   = carve(8, 256);
  _Float16* Wp_lat  = carve(8, 128);
  _Float16* Wp_init = carve(8, 64);
  _Float16* Wp_d1   = carve(3, 256);
  _Float16* Wp_d2   = carve(8, 256);
  _Float16* Wp_d3   = carve(8, 64);
  _Float16* Wp_g1   = carve(3, 256);
  _Float16* Wp_g2   = carve(8, 64);

  auto pack = [&](const float* src, _Float16* dst, int K, int N, int Ktiles) {
    const size_t total = (size_t)Ktiles * (N / 16) * 512;
    const int blocks = (int)((total + 255) / 256);
    hipLaunchKernelGGL(pack_weights_kernel, dim3(blocks), dim3(256), 0, stream,
                       src, dst, K, N, Ktiles);
  };
  pack(W_in, Wp_in, 256, 256, 8);
  for (int b = 0; b < 3; ++b) {
    pack(Wb1 + (size_t)b * 65536, Wp_b1 + (size_t)b * 65536, 256, 256, 8);
    pack(Wb2 + (size_t)b * 65536, Wp_b2 + (size_t)b * 65536, 256, 256, 8);
  }
  pack(W_fo,   Wp_fo,   256, 256, 8);
  pack(W_lat,  Wp_lat,  256, 128, 8);
  pack(W_init, Wp_init, 256, 64,  8);
  pack(Wd1,    Wp_d1,   65,  256, 3);
  pack(Wd2,    Wp_d2,   256, 256, 8);
  pack(Wd3,    Wp_d3,   256, 64,  8);
  pack(Wg1,    Wp_g1,   65,  256, 3);
  pack(Wg2,    Wp_g2,   256, 64,  8);

  float* out    = (float*)d_out;
  float* out_mu = out;
  float* out_s  = out + (size_t)NSDE_BATCH * NSDE_LAT;
  float* out_zp = out + (size_t)2 * NSDE_BATCH * NSDE_LAT;

  const dim3 grid(NSDE_BATCH / ROWS);
  const dim3 block(NTHREADS);
  const size_t smem_bytes =
      (size_t)(2 * ROWS * NSDE_HID + ROWS * TZLD) * sizeof(_Float16);

  hipLaunchKernelGGL(nsde_fused_kernel, grid, block, smem_bytes, stream,
                     x, ts, noise,
                     Wp_in, b_in, Wp_b1, bb1, Wp_b2, bb2, Wp_fo, b_fo,
                     Wp_lat, b_lat, Wp_init, b_init,
                     Wp_d1, bd1, Wp_d2, bd2, Wp_d3, bd3,
                     Wp_g1, bg1, Wp_g2, bg2,
                     out_mu, out_s, out_zp);
}
